// GIN_Net_3607772529427
// MI455X (gfx1250) — matrix operand order repaired and checked by reading, not compiled, not run on hardware
//
#include <hip/hip_runtime.h>

typedef __attribute__((ext_vector_type(16))) _Float16 v16h;
typedef __attribute__((ext_vector_type(8)))  float    v8f;

#define NNODES 20000     // multiple of 16 -> no M tail anywhere
#define NEDGES 320000
#define NGRAPH 64
#define NODE_C 39
#define PRE_C  256
#define EDGE_C 10
#define HID    256
#define MIDC   1024
#define OUTC   128
#define LAYERS 5
#define KCAT   295
#define KCATP  320       // K padded to a multiple of 32 (zero-filled)

// ---------------------------------------------------------------------------
// Pack a [K,N] f32 weight matrix into WMMA B-fragment order, f16:
//   Bp[((kt*tilesN + tn)*32 + lane)*16 + i] = B[(kt*32 + (i>>3)*16 + (lane>>4)*8 + (i&7))*N
//                                               + tn*16 + (lane&15)]
// so each lane of the GEMM reads its whole 16-f16 fragment as one 32B load.
// Rows k >= Ksrc are zero (K padding).
// ---------------------------------------------------------------------------
__global__ __launch_bounds__(256) void pack_b_kernel(
    const float* __restrict__ B, _Float16* __restrict__ Bp,
    int N, int Kp, int Ksrc)
{
    const long long total  = (long long)Kp * N;
    const long long stride = (long long)gridDim.x * 256;
    const int tilesN = N >> 4;
    for (long long t = (long long)blockIdx.x * 256 + threadIdx.x; t < total; t += stride) {
        const int frag = (int)(t >> 9);          // / 512 elements per fragment
        const int r    = (int)(t & 511);
        const int lane = r >> 4;
        const int i    = r & 15;
        const int kt   = frag / tilesN;
        const int tn   = frag - kt * tilesN;
        const int k    = (kt << 5) + ((i >> 3) << 4) + ((lane >> 4) << 3) + (i & 7);
        const int n    = (tn << 4) + (lane & 15);
        const float v  = (k < Ksrc) ? B[(size_t)k * N + n] : 0.0f;
        Bp[t] = (_Float16)v;
    }
}

// ---------------------------------------------------------------------------
// WMMA GEMM: C[M,N] = epi(A (+A2) @ B + bias). One wave -> 16x64 strip:
// one A fragment feeds 4 v_wmma_f32_16x16x32_f16 per k-step.
// Requires M%16==0, N%64==0, K%32==0 (guaranteed by problem shapes+padding),
// so the inner loop is branch-free straight-line code.
// ---------------------------------------------------------------------------
template <bool RELU, bool HASA2, bool POOL>
__global__ __launch_bounds__(256) void wmma_gemm_kernel(
    const float* __restrict__ A, const float* __restrict__ A2,
    const _Float16* __restrict__ Bp, const float* __restrict__ bias,
    float* __restrict__ C, const int* __restrict__ batch,
    int M, int N, int K)
{
    const int lane   = threadIdx.x & 31;
    const int wave   = threadIdx.x >> 5;
    const int tilesN = N >> 4;
    const int macroN = tilesN >> 2;                 // groups of 4 N-tiles
    const int tile   = blockIdx.x * 8 + wave;       // wave-uniform
    if (tile >= (M >> 4) * macroN) return;          // uniform exit, EXEC stays full
    const int tm   = tile / macroN;
    const int tn0  = (tile - tm * macroN) << 2;     // first of 4 N-tiles
    const int row0 = tm << 4;
    const int grp  = lane >> 4;                     // lane half
    const int rA   = row0 + (lane & 15);

    const float* __restrict__ arow  = A + (size_t)rA * K;
    const float* __restrict__ arow2 = HASA2 ? (A2 + (size_t)rA * K) : arow;
    const v16h*  __restrict__ Bv    = reinterpret_cast<const v16h*>(Bp);

    v8f acc[4] = {v8f{}, v8f{}, v8f{}, v8f{}};

    for (int k0 = 0; k0 < K; k0 += 32) {
        // A fragment: two contiguous 8-float runs -> four b128 loads.
        const float4* pa = reinterpret_cast<const float4*>(arow + k0 + (grp << 3));
        const float4* pb = reinterpret_cast<const float4*>(arow + k0 + 16 + (grp << 3));
        float4 q0 = pa[0], q1 = pa[1], q2 = pb[0], q3 = pb[1];
        if (HASA2) {
            const float4* ra = reinterpret_cast<const float4*>(arow2 + k0 + (grp << 3));
            const float4* rb = reinterpret_cast<const float4*>(arow2 + k0 + 16 + (grp << 3));
            float4 s0 = ra[0], s1 = ra[1], s2 = rb[0], s3 = rb[1];
            q0.x += s0.x; q0.y += s0.y; q0.z += s0.z; q0.w += s0.w;
            q1.x += s1.x; q1.y += s1.y; q1.z += s1.z; q1.w += s1.w;
            q2.x += s2.x; q2.y += s2.y; q2.z += s2.z; q2.w += s2.w;
            q3.x += s3.x; q3.y += s3.y; q3.z += s3.z; q3.w += s3.w;
        }
        v16h af;
        af[0]  = (_Float16)q0.x; af[1]  = (_Float16)q0.y;
        af[2]  = (_Float16)q0.z; af[3]  = (_Float16)q0.w;
        af[4]  = (_Float16)q1.x; af[5]  = (_Float16)q1.y;
        af[6]  = (_Float16)q1.z; af[7]  = (_Float16)q1.w;
        af[8]  = (_Float16)q2.x; af[9]  = (_Float16)q2.y;
        af[10] = (_Float16)q2.z; af[11] = (_Float16)q2.w;
        af[12] = (_Float16)q3.x; af[13] = (_Float16)q3.y;
        af[14] = (_Float16)q3.z; af[15] = (_Float16)q3.w;

        // Pre-packed B fragments: one 32B contiguous load per sub-tile.
        const size_t fbase = (size_t)(k0 >> 5) * tilesN;
        v16h b0 = Bv[(fbase + tn0 + 0) * 32 + lane];
        v16h b1 = Bv[(fbase + tn0 + 1) * 32 + lane];
        v16h b2 = Bv[(fbase + tn0 + 2) * 32 + lane];
        v16h b3 = Bv[(fbase + tn0 + 3) * 32 + lane];

        acc[0] = __builtin_amdgcn_wmma_f32_16x16x32_f16(false, af, false, b0,
                                                        (short)0, acc[0], false, false);
        acc[1] = __builtin_amdgcn_wmma_f32_16x16x32_f16(false, af, false, b1,
                                                        (short)0, acc[1], false, false);
        acc[2] = __builtin_amdgcn_wmma_f32_16x16x32_f16(false, af, false, b2,
                                                        (short)0, acc[2], false, false);
        acc[3] = __builtin_amdgcn_wmma_f32_16x16x32_f16(false, af, false, b3,
                                                        (short)0, acc[3], false, false);
    }

    const int mbase = row0 + (grp << 3);            // C layout: elem r -> M=r+grp*8
    #pragma unroll
    for (int t = 0; t < 4; ++t) {
        const int nB = ((tn0 + t) << 4) + (lane & 15);
        const float bv = bias[nB];
        #pragma unroll
        for (int r = 0; r < 8; ++r) {
            const int m = mbase + r;
            float v = acc[t][r] + bv;
            if (RELU) v = fmaxf(v, 0.0f);
            if (POOL) {
                atomicAdd(&C[(size_t)batch[m] * N + nB], v);
            } else {
                C[(size_t)m * N + nB] = v;
            }
        }
    }
}

// ---------------------------------------------------------------------------
// Fused GINE message + scatter: per edge e,
//   msg[c] = relu(h[src][c] + b_edge[c] + sum_k edge_attr[e][k]*W_edge[k][c])
//   agg[dst][c] += msg[c]   (f32 global atomics, L2-resident)
// ---------------------------------------------------------------------------
__global__ __launch_bounds__(256) void edge_message_kernel(
    const float* __restrict__ h, const float* __restrict__ edge_attr,
    const int* __restrict__ src, const int* __restrict__ dst,
    const float* __restrict__ We, const float* __restrict__ be,
    float* __restrict__ agg, int E)
{
    __shared__ float sW[EDGE_C * HID];
    __shared__ float sb[HID];
    for (int i = threadIdx.x; i < EDGE_C * HID; i += 256) sW[i] = We[i];
    for (int i = threadIdx.x; i < HID; i += 256) sb[i] = be[i];
    __syncthreads();

    const int lane   = threadIdx.x & 31;
    const int wave   = threadIdx.x >> 5;
    const int stride = gridDim.x * 8;
    for (int e = blockIdx.x * 8 + wave; e < E; e += stride) {
        const int s = src[e];
        const int d = dst[e];
        float ea[EDGE_C];
        #pragma unroll
        for (int k = 0; k < EDGE_C; ++k) ea[k] = edge_attr[(size_t)e * EDGE_C + k];
        const float* __restrict__ hs = h + (size_t)s * HID;
        float* __restrict__ ad = agg + (size_t)d * HID;
        #pragma unroll
        for (int j = 0; j < HID / 32; ++j) {
            const int c = lane + j * 32;
            float v = sb[c];
            #pragma unroll
            for (int k = 0; k < EDGE_C; ++k) v = fmaf(ea[k], sW[k * HID + c], v);
            v += hs[c];
            v = fmaxf(v, 0.0f);
            atomicAdd(&ad[c], v);
        }
    }
}

// concat(x, pre_x) -> [NNODES, KCATP], zero-padded to K%32==0
__global__ __launch_bounds__(256) void concat_kernel(
    const float* __restrict__ x, const float* __restrict__ pre,
    float* __restrict__ xc)
{
    const long long total  = (long long)NNODES * KCATP;
    const long long stride = (long long)gridDim.x * 256;
    for (long long i = (long long)blockIdx.x * 256 + threadIdx.x; i < total; i += stride) {
        const int row = (int)(i / KCATP);
        const int k   = (int)(i - (long long)row * KCATP);
        float v = 0.0f;
        if (k < NODE_C)     v = x[(size_t)row * NODE_C + k];
        else if (k < KCAT)  v = pre[(size_t)row * PRE_C + (k - NODE_C)];
        xc[i] = v;
    }
}

__global__ __launch_bounds__(256) void zero_kernel(float* __restrict__ p, long long n)
{
    const long long stride = (long long)gridDim.x * 256;
    for (long long i = (long long)blockIdx.x * 256 + threadIdx.x; i < n; i += stride)
        p[i] = 0.0f;
}

static inline int gemm_blocks(int M, int N)
{
    const int tiles = (M / 16) * (N / 64);   // one wave per 16x64 strip
    return (tiles + 7) / 8;
}

extern "C" void kernel_launch(void* const* d_in, const int* in_sizes, int n_in,
                              void* d_out, int out_size, void* d_ws, size_t ws_size,
                              hipStream_t stream)
{
    const float* x         = (const float*)d_in[0];
    const float* pre_x     = (const float*)d_in[1];
    const int*   edge_idx  = (const int*)d_in[2];
    const float* edge_attr = (const float*)d_in[3];
    const int*   batch     = (const int*)d_in[4];
    const float* W_in      = (const float*)d_in[5];
    const float* b_in      = (const float*)d_in[6];
    const float* W_nn      = (const float*)d_in[7];
    const float* b_nn      = (const float*)d_in[8];
    const float* W_edge    = (const float*)d_in[9];   // [L, EDGE_C, HID]
    const float* b_edge    = (const float*)d_in[10];  // [L, HID]
    const float* W1        = (const float*)d_in[11];
    const float* b1        = (const float*)d_in[12];
    const float* W2        = (const float*)d_in[13];
    const float* b2        = (const float*)d_in[14];

    float* ws   = (float*)d_ws;
    float* xcat = ws;                                   // NNODES * KCATP
    float* h_a  = xcat + (size_t)NNODES * KCATP;        // NNODES * HID
    float* h_b  = h_a  + (size_t)NNODES * HID;
    float* agg  = h_b  + (size_t)NNODES * HID;
    float* mid  = agg  + (size_t)NNODES * HID;          // NNODES * MIDC
    _Float16* wp_in = (_Float16*)(mid + (size_t)NNODES * MIDC);
    _Float16* wp_nn = wp_in + (size_t)KCATP * HID;
    _Float16* wp_1  = wp_nn + (size_t)HID * HID;
    _Float16* wp_2  = wp_1  + (size_t)HID * MIDC;

    const int* src = edge_idx;            // edge_index[0]
    const int* dst = edge_idx + NEDGES;   // edge_index[1]

    // Pack weights into WMMA B-fragment order (f16), once per launch.
    pack_b_kernel<<<512, 256, 0, stream>>>(W_in, wp_in, HID,  KCATP, KCAT);
    pack_b_kernel<<<512, 256, 0, stream>>>(W_nn, wp_nn, HID,  HID,   HID);
    pack_b_kernel<<<512, 256, 0, stream>>>(W1,   wp_1,  MIDC, HID,   HID);
    pack_b_kernel<<<512, 256, 0, stream>>>(W2,   wp_2,  OUTC, MIDC,  MIDC);

    // h = concat(x, pre_x) @ W_in + b_in   (no activation)
    concat_kernel<<<4096, 256, 0, stream>>>(x, pre_x, xcat);
    wmma_gemm_kernel<false, false, false>
        <<<gemm_blocks(NNODES, HID), 256, 0, stream>>>(
            xcat, nullptr, wp_in, b_in, h_a, nullptr, NNODES, HID, KCATP);

    float* hc = h_a;
    float* hn = h_b;
    for (int l = 0; l < LAYERS; ++l) {
        zero_kernel<<<2048, 256, 0, stream>>>(agg, (long long)NNODES * HID);
        edge_message_kernel<<<2048, 256, 0, stream>>>(
            hc, edge_attr, src, dst,
            W_edge + (size_t)l * EDGE_C * HID,
            b_edge + (size_t)l * HID, agg, NEDGES);
        // h = relu((h + agg) @ W_nn + b_nn), double-buffered
        wmma_gemm_kernel<true, true, false>
            <<<gemm_blocks(NNODES, HID), 256, 0, stream>>>(
                hc, agg, wp_nn, b_nn, hn, nullptr, NNODES, HID, HID);
        float* t = hc; hc = hn; hn = t;
    }

    // MLP head: mid = relu(h @ W1 + b1); out = pool(mid @ W2 + b2, batch)
    wmma_gemm_kernel<true, false, false>
        <<<gemm_blocks(NNODES, MIDC), 256, 0, stream>>>(
            hc, nullptr, wp_1, b1, mid, nullptr, NNODES, MIDC, HID);
    zero_kernel<<<32, 256, 0, stream>>>((float*)d_out, (long long)NGRAPH * OUTC);
    wmma_gemm_kernel<false, false, true>
        <<<gemm_blocks(NNODES, OUTC), 256, 0, stream>>>(
            mid, nullptr, wp_2, b2, (float*)d_out, batch, NNODES, OUTC, MIDC);
}
